// SequenceDiff_87308095193400
// MI455X (gfx1250) — matrix operand order repaired and checked by reading, not compiled, not run on hardware
//
#include <hip/hip_runtime.h>
#include <hip/hip_bf16.h>
#include <math.h>

// ---------------- problem constants (from reference) ----------------
#define B_      32
#define H_      12
#define S_      512
#define D_      768
#define L_      256      // S/2
#define HH_     384      // hidden per direction
#define G_      1536     // 4*HH_
#define DIN_    1536     // 2*D
#define NSEQ_   (B_*H_)  // 384 sequences through the BiLSTM
#define MROWS_  (NSEQ_*6)// 2304 rows of the input-projection GEMM
#define SLIDER_ 6
#define EPS_    1e-9f

// ---------------- WMMA vector types (gfx1250, wave32) ----------------
typedef __attribute__((ext_vector_type(16))) __bf16 v16bf;
typedef __attribute__((ext_vector_type(8)))  __bf16 v8bf;
typedef __attribute__((ext_vector_type(8)))  float  v8f;

// ---------------- workspace layout (all 256B-aligned by construction) ----
static constexpr size_t OFF_WIN   = 0;                                   // int[2*NSEQ_]
static constexpr size_t SZ_WIN    = 2*NSEQ_*sizeof(int);                 // 3072
static constexpr size_t OFF_SLIDE = OFF_WIN + SZ_WIN;                    // bf16[MROWS_*DIN_]
static constexpr size_t SZ_SLIDE  = (size_t)MROWS_*DIN_*2;
static constexpr size_t OFF_WIH   = OFF_SLIDE + SZ_SLIDE;                // bf16[2][G_*DIN_]
static constexpr size_t SZ_WIH    = (size_t)2*G_*DIN_*2;
static constexpr size_t OFF_WHH   = OFF_WIH + SZ_WIH;                    // bf16[2][G_*HH_]
static constexpr size_t SZ_WHH    = (size_t)2*G_*HH_*2;
static constexpr size_t OFF_XPROJ = OFF_WHH + SZ_WHH;                    // f32[2][MROWS_*G_]
static constexpr size_t SZ_XPROJ  = (size_t)2*MROWS_*G_*4;
static constexpr size_t OFF_GPRE  = OFF_XPROJ + SZ_XPROJ;                // f32[2][NSEQ_*G_]
static constexpr size_t SZ_GPRE   = (size_t)2*NSEQ_*G_*4;
static constexpr size_t OFF_HBF   = OFF_GPRE + SZ_GPRE;                  // bf16[2][NSEQ_*HH_]  (zeroed)
static constexpr size_t SZ_HBF    = (size_t)2*NSEQ_*HH_*2;
static constexpr size_t OFF_C     = OFF_HBF + SZ_HBF;                    // f32[2][NSEQ_*HH_]   (zeroed)
static constexpr size_t SZ_C      = (size_t)2*NSEQ_*HH_*4;
static constexpr size_t OFF_REP   = OFF_C + SZ_C;                        // f32[NSEQ_*D_]       (zeroed)
static constexpr size_t SZ_REP    = (size_t)NSEQ_*D_*4;
static constexpr size_t OFF_FLAG  = OFF_REP + SZ_REP;                    // int[B_*S_]          (zeroed)
static constexpr size_t SZ_FLAG   = (size_t)B_*S_*4;
static constexpr size_t OFF_LREP  = OFF_FLAG + SZ_FLAG;                  // f32[B_*D_]
static constexpr size_t ZERO_BYTES = SZ_HBF + SZ_C + SZ_REP + SZ_FLAG;   // contiguous zero region

// ---------------- utility kernels ----------------
__global__ void k_zero(unsigned int* __restrict__ p, int nwords) {
  for (int i = blockIdx.x*blockDim.x + threadIdx.x; i < nwords; i += gridDim.x*blockDim.x)
    p[i] = 0u;
}

__global__ void k_f32_to_bf16(const float* __restrict__ src, __bf16* __restrict__ dst, int n) {
  int i = blockIdx.x*blockDim.x + threadIdx.x;
  if (i < n) dst[i] = (__bf16)src[i];
}

// ---------------- master position + window start ----------------
// one block per (b,h,half): row-sums of the LxL diagonal block with SEP->EPS,
// then argmax + _window_start.   (201 MB stream -> HBM-bound, ~9us at 23.3 TB/s)
__global__ void k_master(const float* __restrict__ att, const int* __restrict__ sent,
                         int* __restrict__ win) {
  __shared__ float rowsum[L_];
  __shared__ unsigned char sepm[L_];
  int bid  = blockIdx.x;
  int half = bid & 1;
  int bh   = bid >> 1;          // b*H_+h
  int b    = bh / H_;
  int tid  = threadIdx.x;
  if (tid < L_) sepm[tid] = (sent[b*S_ + half*L_ + tid] == 102) ? 1 : 0;
  __syncthreads();

  int lane = tid & 31, wave = tid >> 5;      // 8 waves of 32
  const float* base = att + (((size_t)bh*S_ + half*L_))*S_ + half*L_;
  for (int q = wave; q < L_; q += 8) {
    const float* row = base + (size_t)q*S_;
    float s = 0.f;
    for (int k = lane; k < L_; k += 32)
      s += sepm[k] ? EPS_ : row[k];
    for (int off = 16; off; off >>= 1) s += __shfl_down(s, off, 32);
    if (lane == 0) rowsum[q] = s;
  }
  __syncthreads();

  if (tid == 0) {
    float best = rowsum[0]; int bi = 0;
    for (int q = 1; q < L_; q++) { float v = rowsum[q]; if (v > best) { best = v; bi = q; } }
    int pos = bi;
    if (pos < 1) pos = 1;
    if (pos > L_-2) pos = L_-2;
    int l = SLIDER_/2;
    if (pos - SLIDER_/2 <= 0) l = pos - 1;
    int r = SLIDER_ - l;
    if (pos + r >= L_-1) l = SLIDER_ - (L_ - pos - 2);
    win[half*NSEQ_ + bh] = pos - l;
  }
}

// ---------------- gather slide windows (masked) into bf16 A-matrix ----------------
__global__ void k_gather(const float* __restrict__ logits, const int* __restrict__ mask,
                         const int* __restrict__ win, __bf16* __restrict__ slide) {
  int m   = blockIdx.x;                 // 0..MROWS_-1
  int seq = m / SLIDER_, t = m % SLIDER_;
  int b   = seq / H_;
  int ia  = win[seq] + t;               // A-side absolute position
  int ib  = L_ + win[NSEQ_ + seq] + t;  // B-side absolute position
  bool ma = mask[b*S_ + ia] != 0;
  bool mb = mask[b*S_ + ib] != 0;
  const float* la = logits + ((size_t)b*S_ + ia)*D_;
  const float* lb = logits + ((size_t)b*S_ + ib)*D_;
  __bf16* out = slide + (size_t)m*DIN_;
  for (int d = threadIdx.x; d < D_; d += blockDim.x) {
    out[d]      = (__bf16)(ma ? la[d] : EPS_);
    out[D_ + d] = (__bf16)(mb ? lb[d] : EPS_);
  }
}

__global__ void k_flags(const int* __restrict__ win, int* __restrict__ flag) {
  int seq = threadIdx.x;
  if (seq >= NSEQ_) return;
  int b  = seq / H_;
  int a0 = win[seq];
  int b0 = L_ + win[NSEQ_ + seq];
  for (int t = 0; t < SLIDER_; t++) { flag[b*S_ + a0 + t] = 1; flag[b*S_ + b0 + t] = 1; }
}

// ---------------- register-blocked WMMA macro-tile ------------------------------
// Per wave: 32(M) x 64(N) output, K-loop of 32. 2 A-frags + 4 B-frags -> 8 WMMAs
// per K-step (wmma : b128-load = 8:12 vs 1:4 unblocked).
// A: row-major, tile base already offset to m0. W: row-major [N x K] (GEMM is
// X*W^T so W rows are output columns; contiguous K matches the B-operand lanes).
__device__ __forceinline__ void wmma_tile_2x4(const __bf16* __restrict__ A, int lda,
                                              const __bf16* __restrict__ W, int ldw,
                                              int K, v8f (&c)[2][4]) {
  int lane = threadIdx.x & 31;
  int half = lane >> 4;
  int r    = lane & 15;
  const __bf16* ar0 = A + (size_t)r*lda;          // rows m0 .. m0+15
  const __bf16* ar1 = A + (size_t)(r+16)*lda;     // rows m0+16 .. m0+31
  const __bf16* wr  = W + (size_t)r*ldw;          // cols n0+r (+16j below)
  #pragma unroll 2
  for (int k0 = 0; k0 < K; k0 += 32) {
    // prefetch next K-block of the streamed operand (emits global_prefetch_b8)
    __builtin_prefetch(wr + k0 + 64, 0, 1);
    v8bf lo0 = *(const v8bf*)(ar0 + k0 + half*8);
    v8bf hi0 = *(const v8bf*)(ar0 + k0 + 16 + half*8);
    v8bf lo1 = *(const v8bf*)(ar1 + k0 + half*8);
    v8bf hi1 = *(const v8bf*)(ar1 + k0 + 16 + half*8);
    v16bf a0 = __builtin_shufflevector(lo0, hi0, 0,1,2,3,4,5,6,7,8,9,10,11,12,13,14,15);
    v16bf a1 = __builtin_shufflevector(lo1, hi1, 0,1,2,3,4,5,6,7,8,9,10,11,12,13,14,15);
    #pragma unroll
    for (int j = 0; j < 4; j++) {
      v8bf blo = *(const v8bf*)(wr + (size_t)(16*j)*ldw + k0 + half*8);
      v8bf bhi = *(const v8bf*)(wr + (size_t)(16*j)*ldw + k0 + 16 + half*8);
      v16bf bm = __builtin_shufflevector(blo, bhi, 0,1,2,3,4,5,6,7,8,9,10,11,12,13,14,15);
      c[0][j] = __builtin_amdgcn_wmma_f32_16x16x32_bf16(false, a0, false, bm, (short)0, c[0][j], false, false);
      c[1][j] = __builtin_amdgcn_wmma_f32_16x16x32_bf16(false, a1, false, bm, (short)0, c[1][j], false, false);
    }
  }
}

// ---------------- input projection: Xproj = slide @ wih^T + bih + bhh -------------
// grid: (N/256, M/32, dir), block 128 (4 waves, each a 32x64 macro-tile along N)
__global__ void k_xproj(const __bf16* __restrict__ slide, const __bf16* __restrict__ wih,
                        const float* __restrict__ bih_f, const float* __restrict__ bhh_f,
                        const float* __restrict__ bih_r, const float* __restrict__ bhh_r,
                        float* __restrict__ xproj) {
  int dir  = blockIdx.z;
  int wave = threadIdx.x >> 5, lane = threadIdx.x & 31;
  int n0   = (blockIdx.x*4 + wave)*64;
  int m0   = blockIdx.y*32;
  const __bf16* W = wih + (size_t)dir*G_*DIN_ + (size_t)n0*DIN_;
  const __bf16* A = slide + (size_t)m0*DIN_;
  v8f c[2][4] = {};
  wmma_tile_2x4(A, DIN_, W, DIN_, DIN_, c);
  const float* b0 = dir ? bih_r : bih_f;
  const float* b1 = dir ? bhh_r : bhh_f;
  int half = lane >> 4, col = lane & 15;
  float* out = xproj + (size_t)dir*MROWS_*G_;
  #pragma unroll
  for (int j = 0; j < 4; j++) {
    int n = n0 + 16*j + col;
    float bias = b0[n] + b1[n];
    #pragma unroll
    for (int i = 0; i < 2; i++) {
      #pragma unroll
      for (int e = 0; e < 8; e++) {
        int row = m0 + 16*i + half*8 + e;
        out[(size_t)row*G_ + n] = c[i][j][e] + bias;
      }
    }
  }
}

// ---------------- recurrent GEMM: gpre = Xproj[:,t] + h @ whh^T -------------------
__global__ void k_gates(const __bf16* __restrict__ hbf, const __bf16* __restrict__ whh,
                        const float* __restrict__ xproj, float* __restrict__ gpre, int step) {
  int dir  = blockIdx.z;
  int wave = threadIdx.x >> 5, lane = threadIdx.x & 31;
  int n0   = (blockIdx.x*4 + wave)*64;
  int m0   = blockIdx.y*32;
  const __bf16* A = hbf + (size_t)dir*NSEQ_*HH_ + (size_t)m0*HH_;
  const __bf16* W = whh + (size_t)dir*G_*HH_   + (size_t)n0*HH_;
  v8f c[2][4] = {};
  wmma_tile_2x4(A, HH_, W, HH_, HH_, c);
  int tt = dir ? (SLIDER_-1-step) : step;   // bwd scan runs the sequence reversed
  const float* xp = xproj + (size_t)dir*MROWS_*G_;
  float* out = gpre + (size_t)dir*NSEQ_*G_;
  int half = lane >> 4, col = lane & 15;
  #pragma unroll
  for (int j = 0; j < 4; j++) {
    int n = n0 + 16*j + col;
    #pragma unroll
    for (int i = 0; i < 2; i++) {
      #pragma unroll
      for (int e = 0; e < 8; e++) {
        int row = m0 + 16*i + half*8 + e;
        out[(size_t)row*G_ + n] = c[i][j][e] + xp[((size_t)row*SLIDER_ + tt)*G_ + n];
      }
    }
  }
}

// ---------------- LSTM pointwise: gates -> (c,h); accumulate rep; h -> bf16 -------
__global__ void k_lstm_pw(const float* __restrict__ gpre, float* __restrict__ cst,
                          __bf16* __restrict__ hbf, float* __restrict__ rep) {
  int idx = blockIdx.x*blockDim.x + threadIdx.x;
  if (idx >= 2*NSEQ_*HH_) return;
  int dir = idx / (NSEQ_*HH_);
  int rem = idx - dir*(NSEQ_*HH_);
  int seq = rem / HH_;
  int j   = rem - seq*HH_;
  const float* g = gpre + (size_t)dir*NSEQ_*G_ + (size_t)seq*G_;
  float i_ = g[j], f_ = g[HH_+j], gg = g[2*HH_+j], o_ = g[3*HH_+j];
  float c = cst[(size_t)dir*NSEQ_*HH_ + rem];
  float si = 1.f/(1.f+__expf(-i_));
  float sf = 1.f/(1.f+__expf(-f_));
  float so = 1.f/(1.f+__expf(-o_));
  c = sf*c + si*tanhf(gg);
  float h = so*tanhf(c);
  cst[(size_t)dir*NSEQ_*HH_ + rem] = c;
  hbf[(size_t)dir*NSEQ_*HH_ + rem] = (__bf16)h;
  rep[(size_t)seq*D_ + dir*HH_ + j] += h;
}

// ---------------- logits_rep: mean over S with flagged/masked substitution --------
__global__ void k_lrep(const float* __restrict__ logits, const int* __restrict__ mask,
                       const int* __restrict__ flag, float* __restrict__ lrep) {
  int b = blockIdx.x;
  int d = blockIdx.y*blockDim.x + threadIdx.x;
  if (d >= D_) return;
  float acc = 0.f;
  for (int s = 0; s < S_; s++) {
    float v = logits[((size_t)b*S_ + s)*D_ + d];
    if (flag[b*S_+s] && mask[b*S_+s] == 0) v = EPS_;   // idempotent .at[].set()
    acc += v;
  }
  lrep[b*D_ + d] = acc * (1.f/S_);
}

// ---------------- final FC + softmax over 2 classes -------------------------------
__global__ void k_fc(const float* __restrict__ rep, const float* __restrict__ lrep,
                     const float* __restrict__ fcw, const float* __restrict__ fcb,
                     float* __restrict__ out) {
  const int NX = D_*(H_+1);   // 9984
  int b = blockIdx.x, tid = threadIdx.x;
  float p0 = 0.f, p1 = 0.f;
  for (int i = tid; i < NX; i += blockDim.x) {
    float xv;
    if (i < H_*D_) xv = rep[(size_t)b*H_*D_ + i] * (1.f/SLIDER_);  // mean over 6 steps
    else           xv = lrep[b*D_ + (i - H_*D_)];
    p0 += xv*fcw[i]; p1 += xv*fcw[NX+i];
  }
  __shared__ float s0[256], s1[256];
  s0[tid] = p0; s1[tid] = p1; __syncthreads();
  for (int o = 128; o; o >>= 1) {
    if (tid < o) { s0[tid] += s0[tid+o]; s1[tid] += s1[tid+o]; }
    __syncthreads();
  }
  if (tid == 0) {
    float z0 = s0[0] + fcb[0], z1 = s1[0] + fcb[1];
    float m = fmaxf(z0, z1);
    float e0 = __expf(z0-m), e1 = __expf(z1-m), s = e0+e1;
    out[b*2+0] = e0/s; out[b*2+1] = e1/s;
  }
}

// ---------------- host launcher ----------------
extern "C" void kernel_launch(void* const* d_in, const int* in_sizes, int n_in,
                              void* d_out, int out_size, void* d_ws, size_t ws_size,
                              hipStream_t stream) {
  const int*   sent   = (const int*)  d_in[0];
  const float* att    = (const float*)d_in[1];
  const float* logits = (const float*)d_in[2];
  const int*   mask   = (const int*)  d_in[3];
  const float* wih_f  = (const float*)d_in[4];
  const float* whh_f  = (const float*)d_in[5];
  const float* bih_f  = (const float*)d_in[6];
  const float* bhh_f  = (const float*)d_in[7];
  const float* wih_r  = (const float*)d_in[8];
  const float* whh_r  = (const float*)d_in[9];
  const float* bih_r  = (const float*)d_in[10];
  const float* bhh_r  = (const float*)d_in[11];
  const float* fc_w   = (const float*)d_in[12];
  const float* fc_b   = (const float*)d_in[13];
  float* out = (float*)d_out;
  char*  ws  = (char*)d_ws;

  int*    win   = (int*)   (ws + OFF_WIN);
  __bf16* slide = (__bf16*)(ws + OFF_SLIDE);
  __bf16* wihb  = (__bf16*)(ws + OFF_WIH);
  __bf16* whhb  = (__bf16*)(ws + OFF_WHH);
  float*  xproj = (float*) (ws + OFF_XPROJ);
  float*  gpre  = (float*) (ws + OFF_GPRE);
  __bf16* hbf   = (__bf16*)(ws + OFF_HBF);
  float*  cst   = (float*) (ws + OFF_C);
  float*  rep   = (float*) (ws + OFF_REP);
  int*    flag  = (int*)   (ws + OFF_FLAG);
  float*  lrep  = (float*) (ws + OFF_LREP);

  // 0) zero recurrent state / rep / flags (one contiguous region)
  {
    int nwords = (int)(ZERO_BYTES/4);
    k_zero<<<dim3((nwords+255)/256), dim3(256), 0, stream>>>((unsigned int*)(ws+OFF_HBF), nwords);
  }
  // 1) weights -> bf16
  {
    int n1 = G_*DIN_, n2 = G_*HH_;
    k_f32_to_bf16<<<dim3(n1/256), dim3(256), 0, stream>>>(wih_f, wihb,              n1);
    k_f32_to_bf16<<<dim3(n1/256), dim3(256), 0, stream>>>(wih_r, wihb + (size_t)n1, n1);
    k_f32_to_bf16<<<dim3(n2/256), dim3(256), 0, stream>>>(whh_f, whhb,              n2);
    k_f32_to_bf16<<<dim3(n2/256), dim3(256), 0, stream>>>(whh_r, whhb + (size_t)n2, n2);
  }
  // 2) attention block sums -> argmax -> window starts   (201 MB stream, HBM-bound)
  k_master<<<dim3(NSEQ_*2), dim3(256), 0, stream>>>(att, sent, win);
  // 3) gather masked slide windows (bf16) + modified-position flags
  k_gather<<<dim3(MROWS_), dim3(256), 0, stream>>>(logits, mask, win, slide);
  k_flags <<<dim3(1), dim3(NSEQ_), 0, stream>>>(win, flag);
  // 4) input projection GEMM (WMMA bf16): M=2304 N=1536 K=1536 x 2 dirs
  k_xproj<<<dim3(G_/256, MROWS_/32, 2), dim3(128), 0, stream>>>(
      slide, wihb, bih_f, bhh_f, bih_r, bhh_r, xproj);
  // 5) six recurrent steps (WMMA bf16 GEMM + pointwise LSTM cell)
  for (int t = 0; t < SLIDER_; t++) {
    k_gates  <<<dim3(G_/256, NSEQ_/32, 2), dim3(128), 0, stream>>>(hbf, whhb, xproj, gpre, t);
    k_lstm_pw<<<dim3((2*NSEQ_*HH_+255)/256), dim3(256), 0, stream>>>(gpre, cst, hbf, rep);
  }
  // 6) logits_rep with idempotent substitution folded into the mean
  k_lrep<<<dim3(B_, (D_+127)/128), dim3(128), 0, stream>>>(logits, mask, flag, lrep);
  // 7) FC + softmax
  k_fc<<<dim3(B_), dim3(256), 0, stream>>>(rep, lrep, fc_w, fc_b, out);
}
